// WindowAttention_83554293776781
// MI455X (gfx1250) — compile-verified, matrix operand order checked by reading, not run on hardware
//
#include <hip/hip_runtime.h>
#include <hip/hip_bf16.h>
#include <math.h>

// ---------------------------------------------------------------------------
// Swin-V2 window attention, fused per-window kernel for gfx1250 (CDNA5).
// One block = one window (L=64, D=256), 8 waves (wave32) = 8 heads.
// All GEMMs via v_wmma_f32_16x16x32_f16, f32 accumulation.
// Weights pre-converted to f16 in workspace (loaded as b128, no cvt in loop).
// ---------------------------------------------------------------------------

typedef __attribute__((ext_vector_type(16))) _Float16 v16h;
typedef __attribute__((ext_vector_type(8)))  _Float16 v8h;
typedef __attribute__((ext_vector_type(8)))  float    v8f;

#define WS   8
#define NH   8
#define DIM  256
#define HD   32
#define L    64
#define NW   64

#define LDX 264   // x / O staging stride (halfs), 528B = 33*16B -> 16B aligned rows
#define LDQ 776   // qkv stride (halfs),          1552B = 97*16B -> 16B aligned rows
#define LDP 72    // P tile stride (halfs),        144B =  9*16B -> 16B aligned rows

// ---------------------------------------------------------------------------
// Fragment loaders (ISA §7.12.2 layouts).
// A/B 16-bit fragment: lane<16 -> row/col row0+lane, halfs 0..7 = K0+{0..7},
// halfs 8..15 = K0+{16..23}; lanes 16..31 same rows, K chunks shifted by 8.
// Works for f16 LDS (A side) and row-major f16 weights W[N][K] (B = W^T).
// ---------------------------------------------------------------------------
__device__ __forceinline__ v16h frag_h(const _Float16* S, int ld, int row0,
                                       int k0, int lane) {
  const int l16 = lane & 15, hf = lane >> 4;
  const _Float16* p = S + (size_t)(row0 + l16) * ld + k0 + hf * 8;
  v8h lo = *(const v8h*)p;          // 16B aligned: ld, k0, hf*8 all mult. of 8
  v8h hi = *(const v8h*)(p + 16);
  return __builtin_shufflevector(lo, hi, 0, 1, 2, 3, 4, 5, 6, 7,
                                 8, 9, 10, 11, 12, 13, 14, 15);
}

// Scaled fragment from f16 LDS: row r multiplied by rs[r] (cosine-attn norms).
__device__ __forceinline__ v16h frag_scaled(const _Float16* S, int ld, int row0,
                                            int col0, const float* rs, int lane) {
  const int l16 = lane & 15, hf = lane >> 4;
  const int r = row0 + l16;
  const float sc = rs[r];
  const _Float16* p = S + r * ld + col0 + hf * 8;
  v16h f;
#pragma unroll
  for (int i = 0; i < 8; ++i) {
    f[i]     = (_Float16)((float)p[i] * sc);
    f[8 + i] = (_Float16)((float)p[16 + i] * sc);
  }
  return f;
}

// B fragment for P@V: B[k][d] = V[k][d] stored row-major in LDS (column reads).
__device__ __forceinline__ v16h frag_vcol(const _Float16* S, int ld, int k0,
                                          int col0, int lane) {
  const int l16 = lane & 15, hf = lane >> 4;
  const _Float16* p = S + (k0 + hf * 8) * ld + col0 + l16;
  v16h f;
#pragma unroll
  for (int i = 0; i < 8; ++i) {
    f[i]     = p[i * ld];
    f[8 + i] = p[(16 + i) * ld];
  }
  return f;
}

// ---------------------------------------------------------------------------
// Kernel 0a: CPB-MLP -> 16*sigmoid(bias) table [NH][64][64] in workspace.
// ---------------------------------------------------------------------------
__global__ __launch_bounds__(256) void swin_cpb_kernel(
    const float* __restrict__ w1, const float* __restrict__ b1,
    const float* __restrict__ w2, const float* __restrict__ b2,
    float* __restrict__ bias16) {
  __shared__ float bt[225 * NH];
  const int tid = threadIdx.x;

  if (tid < 225) {
    const int i = tid / 15, j = tid % 15;
    float t0 = (float)(i - 7) * (8.0f / 7.0f);
    float t1 = (float)(j - 7) * (8.0f / 7.0f);
    const float inv_l8 = 1.0f / 3.0f;  // 1/log2(8)
    t0 = (t0 < 0.f ? -1.f : 1.f) * log2f(fabsf(t0) + 1.f) * inv_l8;
    t1 = (t1 < 0.f ? -1.f : 1.f) * log2f(fabsf(t1) + 1.f) * inv_l8;
    float acc[NH];
#pragma unroll
    for (int h = 0; h < NH; ++h) acc[h] = b2[h];
    for (int k = 0; k < 512; ++k) {
      const float hid = fmaxf(t0 * w1[2 * k] + t1 * w1[2 * k + 1] + b1[k], 0.f);
#pragma unroll
      for (int h = 0; h < NH; ++h) acc[h] += w2[h * 512 + k] * hid;
    }
#pragma unroll
    for (int h = 0; h < NH; ++h) bt[tid * NH + h] = acc[h];
  }
  __syncthreads();

  for (int e = tid; e < NH * L * L; e += 256) {
    const int h = e >> 12, m = (e >> 6) & 63, n = e & 63;
    const int ry = (m >> 3) - (n >> 3) + (WS - 1);
    const int rx = (m & 7) - (n & 7) + (WS - 1);
    const float bv = bt[(ry * 15 + rx) * NH + h];
    bias16[e] = 16.f / (1.f + __expf(-bv));
  }
}

// ---------------------------------------------------------------------------
// Kernel 0b: convert GEMM weights to f16 (done once; reused by all windows).
// ---------------------------------------------------------------------------
__global__ __launch_bounds__(256) void swin_wcvt_kernel(
    const float* __restrict__ qkv_w, const float* __restrict__ proj_w,
    _Float16* __restrict__ qw_h, _Float16* __restrict__ pw_h) {
  const int i = blockIdx.x * 256 + threadIdx.x;
  const int stride = gridDim.x * 256;
  for (int e = i; e < 3 * DIM * DIM; e += stride) qw_h[e] = (_Float16)qkv_w[e];
  for (int e = i; e < DIM * DIM; e += stride)     pw_h[e] = (_Float16)proj_w[e];
}

// ---------------------------------------------------------------------------
// Main fused kernel.
// ---------------------------------------------------------------------------
__global__ __launch_bounds__(256) void swin_attn_kernel(
    const float* __restrict__ x, const float* __restrict__ mask,
    const _Float16* __restrict__ qw_h, const float* __restrict__ qkv_b,
    const _Float16* __restrict__ pw_h, const float* __restrict__ proj_b,
    const float* __restrict__ logit_scale, const float* __restrict__ bias16,
    float* __restrict__ out) {
  __shared__ _Float16 xs[L * LDX];        // x (f16), later reused for O
  __shared__ _Float16 qk[L * LDQ];        // qkv (f16): q | k | v per 256 cols
  __shared__ float    rq[NH * L];         // scale/||q|| per (head,row)
  __shared__ float    rk[NH * L];         // 1/||k||    per (head,row)
  __shared__ _Float16 pb[NH * 16 * LDP];  // per-wave 16x64 P tile

  const int b    = blockIdx.x;
  const int tid  = threadIdx.x;
  const int lane = tid & 31;
  const int wv   = tid >> 5;
  const int l16  = lane & 15;
  const int lh   = lane >> 4;

  // Phase 1: stage x -> f16 LDS (vectorized: b128 in, b64 out)
  const float4* xb4 = (const float4*)(x + (size_t)b * L * DIM);
  for (int e = tid; e < L * DIM / 4; e += 256) {
    const float4 v = xb4[e];
    _Float16* d = xs + (e >> 6) * LDX + (e & 63) * 4;
    d[0] = (_Float16)v.x; d[1] = (_Float16)v.y;
    d[2] = (_Float16)v.z; d[3] = (_Float16)v.w;
  }
  __syncthreads();

  // Phase 2: qkv = x @ qkv_w^T + qkv_b   (M=64, N=768, K=256)
  // A fragments hoisted per 16-row block and reused across all N tiles.
  for (int mi = 0; mi < 4; ++mi) {
    v16h a[8];
#pragma unroll
    for (int k0 = 0; k0 < 8; ++k0) a[k0] = frag_h(xs, LDX, mi * 16, k0 * 32, lane);
    for (int ni = wv; ni < 48; ni += 8) {
      if (ni + 8 < 48)
        __builtin_prefetch(qw_h + (size_t)(ni + 8) * 16 * DIM + l16 * DIM, 0, 1);
      v8f c = {};
#pragma unroll
      for (int k0 = 0; k0 < 8; ++k0) {
        v16h bf = frag_h(qw_h, DIM, ni * 16, k0 * 32, lane);
        c = __builtin_amdgcn_wmma_f32_16x16x32_f16(false, a[k0], false, bf,
                                                   (short)0, c, false, false);
      }
      const int n = ni * 16 + l16;
      const float bb = qkv_b[n];
#pragma unroll
      for (int i = 0; i < 8; ++i) {
        const int m = mi * 16 + i + 8 * lh;
        qk[m * LDQ + n] = (_Float16)(c[i] + bb);
      }
    }
  }
  __syncthreads();

  // Phase 2.5: per-(head,row) inverse L2 norms; fold logit scale into q
  for (int e = tid; e < 2 * NH * L; e += 256) {
    const int isk = e >> 9, rr = e & 511;
    const int h = rr >> 6, m = rr & 63;
    const _Float16* p = qk + m * LDQ + isk * DIM + h * HD;
    float s = 0.f;
#pragma unroll
    for (int i = 0; i < HD; ++i) { const float v = (float)p[i]; s += v * v; }
    const float inv = 1.f / fmaxf(sqrtf(s), 1e-12f);
    if (isk) {
      rk[h * L + m] = inv;
    } else {
      const float sc = __expf(fminf(logit_scale[h], 4.6051702f));  // ln(100)
      rq[h * L + m] = inv * sc;
    }
  }
  __syncthreads();

  // Phase 3+4: attention for head h = wv (entirely within one wave)
  {
    const int h = wv;
    const float* bmh = bias16 + (size_t)h * L * L;
    const float* mkh = mask + (size_t)(b & (NW - 1)) * L * L;
    _Float16* pw = pb + h * 16 * LDP;

    // hoist V fragments (B side of P@V): [k-step][d-tile]
    v16h vf[2][2];
#pragma unroll
    for (int k0 = 0; k0 < 2; ++k0)
#pragma unroll
      for (int di = 0; di < 2; ++di)
        vf[k0][di] = frag_vcol(qk, LDQ, k0 * 32, 2 * DIM + h * HD + di * 16, lane);

    for (int mi = 0; mi < 4; ++mi) {
      // S = (q*scale/||q||) @ (k/||k||)^T  — K = 32 -> one WMMA per tile
      v16h a = frag_scaled(qk, LDQ, mi * 16, h * HD, rq + h * L, lane);
      v8f s[4];
#pragma unroll
      for (int ni = 0; ni < 4; ++ni) {
        v16h bf = frag_scaled(qk, LDQ, ni * 16, DIM + h * HD, rk + h * L, lane);
        v8f z = {};
        s[ni] = __builtin_amdgcn_wmma_f32_16x16x32_f16(false, a, false, bf,
                                                       (short)0, z, false, false);
      }
      // + 16*sigmoid(cpb bias) + window mask
#pragma unroll
      for (int ni = 0; ni < 4; ++ni)
#pragma unroll
        for (int i = 0; i < 8; ++i) {
          const int m = mi * 16 + i + 8 * lh, n = ni * 16 + l16;
          s[ni][i] += bmh[m * L + n] + mkh[m * L + n];
        }
      // softmax over n (row lives in 16 lanes of one half + 4 ni tiles)
#pragma unroll
      for (int i = 0; i < 8; ++i) {
        float rm = fmaxf(fmaxf(s[0][i], s[1][i]), fmaxf(s[2][i], s[3][i]));
#pragma unroll
        for (int d = 1; d < 16; d <<= 1) rm = fmaxf(rm, __shfl_xor(rm, d, 32));
        float sum = 0.f;
#pragma unroll
        for (int ni = 0; ni < 4; ++ni) {
          s[ni][i] = __expf(s[ni][i] - rm);
          sum += s[ni][i];
        }
#pragma unroll
        for (int d = 1; d < 16; d <<= 1) sum += __shfl_xor(sum, d, 32);
        const float inv = 1.f / sum;
#pragma unroll
        for (int ni = 0; ni < 4; ++ni) s[ni][i] *= inv;
      }
      // D-layout -> A-layout via per-wave 16x64 LDS tile
#pragma unroll
      for (int ni = 0; ni < 4; ++ni)
#pragma unroll
        for (int i = 0; i < 8; ++i)
          pw[(i + 8 * lh) * LDP + ni * 16 + l16] = (_Float16)s[ni][i];

      // O[mi] = P[mi] @ V   (K = 64), write into xs (x no longer needed)
#pragma unroll
      for (int di = 0; di < 2; ++di) {
        v8f o = {};
#pragma unroll
        for (int k0 = 0; k0 < 2; ++k0) {
          v16h pa = frag_h(pw, LDP, 0, k0 * 32, lane);
          o = __builtin_amdgcn_wmma_f32_16x16x32_f16(false, pa, false, vf[k0][di],
                                                     (short)0, o, false, false);
        }
#pragma unroll
        for (int i = 0; i < 8; ++i) {
          const int m = mi * 16 + i + 8 * lh;
          xs[m * LDX + h * HD + di * 16 + l16] = (_Float16)o[i];
        }
      }
    }
  }
  __syncthreads();

  // Phase 5: out = O @ proj_w^T + proj_b   (M=64, N=256, K=256)
  float* ob = out + (size_t)b * L * DIM;
  for (int mi = 0; mi < 4; ++mi) {
    v16h a[8];
#pragma unroll
    for (int k0 = 0; k0 < 8; ++k0) a[k0] = frag_h(xs, LDX, mi * 16, k0 * 32, lane);
    for (int ni = wv; ni < 16; ni += 8) {
      v8f c = {};
#pragma unroll
      for (int k0 = 0; k0 < 8; ++k0) {
        v16h bf = frag_h(pw_h, DIM, ni * 16, k0 * 32, lane);
        c = __builtin_amdgcn_wmma_f32_16x16x32_f16(false, a[k0], false, bf,
                                                   (short)0, c, false, false);
      }
      const int n = ni * 16 + l16;
      const float pbv = proj_b[n];
#pragma unroll
      for (int i = 0; i < 8; ++i) {
        const int m = mi * 16 + i + 8 * lh;
        ob[m * DIM + n] = c[i] + pbv;
      }
    }
  }
}

// ---------------------------------------------------------------------------
extern "C" void kernel_launch(void* const* d_in, const int* in_sizes, int n_in,
                              void* d_out, int out_size, void* d_ws,
                              size_t ws_size, hipStream_t stream) {
  const float* x      = (const float*)d_in[0];
  const float* mask   = (const float*)d_in[1];
  const float* qkv_w  = (const float*)d_in[2];
  const float* qkv_b  = (const float*)d_in[3];
  const float* proj_w = (const float*)d_in[4];
  const float* proj_b = (const float*)d_in[5];
  const float* ls     = (const float*)d_in[6];
  const float* cw1    = (const float*)d_in[7];
  const float* cb1    = (const float*)d_in[8];
  const float* cw2    = (const float*)d_in[9];
  const float* cb2    = (const float*)d_in[10];
  float* out = (float*)d_out;

  // Workspace layout (640 KB total):
  //   [0, 128K)      bias16  : NH*64*64 f32
  //   [128K, 512K)   qw_h    : 768*256 f16
  //   [512K, 640K)   pw_h    : 256*256 f16
  float*    bias16 = (float*)d_ws;
  _Float16* qw_h   = (_Float16*)((char*)d_ws + 131072);
  _Float16* pw_h   = (_Float16*)((char*)d_ws + 131072 + 393216);

  swin_cpb_kernel<<<1, 256, 0, stream>>>(cw1, cb1, cw2, cb2, bias16);
  swin_wcvt_kernel<<<64, 256, 0, stream>>>(qkv_w, proj_w, qw_h, pw_h);

  const int num_windows = in_sizes[0] / (L * DIM);  // 2048
  swin_attn_kernel<<<num_windows, 256, 0, stream>>>(
      x, mask, qw_h, qkv_b, pw_h, proj_b, ls, bias16, out);
}
